// TemporalWindowAttention_53145925320754
// MI455X (gfx1250) — matrix operand order repaired and checked by reading, not compiled, not run on hardware
//
#include <hip/hip_runtime.h>
#include <math.h>

// ---------------- problem constants ----------------
#define B_      2
#define N_      4
#define C_      256
#define CT_     256
#define H_      96
#define W_DIM   96
#define WS_     8
#define NH_     8
#define NP_     4
#define HD_     (C_ / NH_)          // 32
#define HWIN_   (H_ / WS_)          // 12
#define NUMWIN_ (HWIN_ * HWIN_)     // 144
#define BW_     (B_ * NUMWIN_)      // 288
#define NPIX_Q  (B_ * H_ * W_DIM)   // 18432  == BW_*64
#define NPIX_V  (B_ * N_ * H_ * W_DIM) // 73728
#define GAIN_   1.0f
#define MOS_    2.0f

typedef __attribute__((ext_vector_type(16))) _Float16 v16h;
typedef __attribute__((ext_vector_type(8)))  _Float16 v8h;
typedef __attribute__((ext_vector_type(8)))  float    v8f;
typedef __attribute__((ext_vector_type(4)))  float    v4f;

__device__ __forceinline__ float silu_f(float x) {
    return x / (1.0f + __expf(-x));
}
__device__ __forceinline__ float gelu_tanh(float x) {
    const float k0 = 0.7978845608028654f; // sqrt(2/pi)
    float x3 = x * x * x;
    return 0.5f * x * (1.0f + tanhf(k0 * (x + 0.044715f * x3)));
}

// ---------------------------------------------------------------------------
// Kernel 0: one-time f32 -> f16 weight conversion.
// ---------------------------------------------------------------------------
__global__ void cvt_w_kernel(const float* __restrict__ src,
                             _Float16* __restrict__ dst, int n) {
    int gid = blockIdx.x * blockDim.x + threadIdx.x;
    if (gid < n) dst[gid] = (_Float16)src[gid];
}

// ---------------------------------------------------------------------------
// Kernel 1: AdaLN parameter GEMMs (tiny) + motion scale + dt bias
// ---------------------------------------------------------------------------
__global__ void params_kernel(const float* __restrict__ tgt,
                              const float* __restrict__ Wq, const float* __restrict__ bq,
                              const float* __restrict__ Wkv, const float* __restrict__ bkv,
                              const float* __restrict__ rel_dt,
                              const float* __restrict__ Wms, const float* __restrict__ bms,
                              float* __restrict__ p_q, float* __restrict__ p_kv,
                              float* __restrict__ ms, float* __restrict__ dtb) {
    int gid = blockIdx.x * blockDim.x + threadIdx.x;
    const int PTOT = B_ * 3 * C_;      // 1536 per projection
    if (gid < 2 * PTOT) {
        int which = gid / PTOT;
        int r = gid % PTOT;
        int b = r / (3 * C_), o = r % (3 * C_);
        const float* Wp = which ? Wkv : Wq;
        const float* bp = which ? bkv : bq;
        float acc = bp[o];
        for (int k = 0; k < CT_; ++k) {
            acc += silu_f(tgt[b * CT_ + k]) * Wp[k * (3 * C_) + o];
        }
        (which ? p_kv : p_q)[b * 3 * C_ + o] = acc;
    }
    if (gid < B_ * N_) {
        float da = fabsf(rel_dt[gid]);
        dtb[gid] = -GAIN_ * da;
        for (int h = 0; h < NH_; ++h) {
            float z = da * Wms[h] + bms[h];
            float sg = 1.0f / (1.0f + __expf(-z));
            ms[gid * NH_ + h] = 1.0f + (MOS_ - 1.0f) * sg;
        }
    }
}

// ---------------------------------------------------------------------------
// Kernel 2: AdaLN (channel LN + FiLM) fused with window partition -> f16
// activation matrix (row channel-contiguous, ready for WMMA A-fragments).
// ---------------------------------------------------------------------------
__global__ void adaln_q_kernel(const float* __restrict__ q,
                               const float* __restrict__ p_q,
                               _Float16* __restrict__ Xq) {
    int gid = blockIdx.x * blockDim.x + threadIdx.x;
    if (gid >= NPIX_Q) return;
    int b = gid / (H_ * W_DIM);
    int pix = gid % (H_ * W_DIM);
    int h = pix / W_DIM, w = pix % W_DIM;
    const float* xp = q + (size_t)b * C_ * H_ * W_DIM + pix;
    float mu = 0.f, m2 = 0.f;
    for (int c = 0; c < C_; ++c) {
        float v = xp[(size_t)c * H_ * W_DIM];
        mu += v; m2 += v * v;
    }
    mu *= (1.0f / C_);
    m2 = m2 * (1.0f / C_) - mu * mu;
    float rs = rsqrtf(m2 + 1e-5f);
    int row = ((b * NUMWIN_ + (h >> 3) * HWIN_ + (w >> 3)) << 6) + ((h & 7) << 3) + (w & 7);
    _Float16* xr = Xq + (size_t)row * C_;
    const float* g  = p_q + b * 3 * C_;
    const float* bt = g + C_;
    for (int c = 0; c < C_; ++c) {
        float xn = (xp[(size_t)c * H_ * W_DIM] - mu) * rs;
        xr[c] = (_Float16)(g[c] * xn + bt[c]);
    }
}

__global__ void adaln_v_kernel(const float* __restrict__ vals,
                               const float* __restrict__ p_kv,
                               _Float16* __restrict__ Xv) {
    int gid = blockIdx.x * blockDim.x + threadIdx.x;
    if (gid >= NPIX_V) return;
    int b = gid / (N_ * H_ * W_DIM);
    int rem = gid % (N_ * H_ * W_DIM);
    int n = rem / (H_ * W_DIM);
    int pix = rem % (H_ * W_DIM);
    int h = pix / W_DIM, w = pix % W_DIM;
    const float* xp = vals + ((size_t)(b * N_ + n) * C_) * H_ * W_DIM + pix;
    float mu = 0.f, m2 = 0.f;
    for (int c = 0; c < C_; ++c) {
        float v = xp[(size_t)c * H_ * W_DIM];
        mu += v; m2 += v * v;
    }
    mu *= (1.0f / C_);
    m2 = m2 * (1.0f / C_) - mu * mu;
    float rs = rsqrtf(m2 + 1e-5f);
    int win = (h >> 3) * HWIN_ + (w >> 3);
    int row = (((b * N_ + n) * NUMWIN_ + win) << 6) + ((h & 7) << 3) + (w & 7);
    _Float16* xr = Xv + (size_t)row * C_;
    const float* g  = p_kv + b * 3 * C_;
    const float* bt = g + C_;
    for (int c = 0; c < C_; ++c) {
        float xn = (xp[(size_t)c * H_ * W_DIM] - mu) * rs;
        xr[c] = (_Float16)(g[c] * xn + bt[c]);
    }
}

// ---------------------------------------------------------------------------
// Kernel 3a: LDS-staged, double-buffered WMMA GEMM for the big GEMMs.
// Block = NW waves; block owns a 32-row strip and ALL Nc (= 64*NW) columns.
// Per K-step a 32x32 f16 A tile (2 KB) is staged into the ping-pong LDS
// buffer with global_load_async_to_lds_b128 (ASYNCcnt) *one iteration ahead*,
// overlapping the DMA with the WMMA chain. Each wave reads two A fragments
// from LDS (ds_load_b128) and batches 4 B fragments, issuing 8 back-to-back
// v_wmma_f32_16x16x32_f16 per K-step (each B fragment reused by 2 WMMAs).
// EPI: 0 = f32 + bias ; 1 = f16 + bias + GELU ; 2 = f16 + bias
// ---------------------------------------------------------------------------
template <int NW, int EPI>
__global__ void __launch_bounds__(NW * 32)
wmma_gemm_lds(const _Float16* __restrict__ X,
              const _Float16* __restrict__ W16,   // [Nc, K] f16
              const float* __restrict__ bias,
              float* __restrict__ Yf,
              _Float16* __restrict__ Yh,
              int M, int Nc, int K) {
    __shared__ __align__(16) _Float16 sA[2][32][32];   // ping-pong 32x32 A tiles
    const int tid = threadIdx.x;
    const int wv = tid >> 5, lane = tid & 31;
    const int tm = blockIdx.x;                  // 32-row strip index
    const int kg = lane >> 4;
    const int tn0 = wv * 4;                     // 4 column tiles per wave

    v8f acc[2][4] = {};

    const unsigned sbase = (unsigned)(size_t)&sA[0][0][0];
    const _Float16* Xbase = X + (size_t)tm * 32 * K;

    // stage one 32x32 tile (128 chunks of 16B) into LDS buffer `buf`
    auto stage = [&](int buf, int k0) {
#pragma unroll
        for (int cc = tid; cc < 128; cc += NW * 32) {
            const _Float16* gp = Xbase + (size_t)(cc >> 2) * K + k0 + (cc & 3) * 8;
            unsigned lo = sbase + (unsigned)(buf * 2048 + cc * 16);
            asm volatile("global_load_async_to_lds_b128 %0, %1, off"
                         :: "v"(lo), "v"(gp) : "memory");
        }
    };

    stage(0, 0);   // prologue: first tile in flight

    const _Float16* pb = W16 + (size_t)(tn0 * 16 + (lane & 15)) * K + kg * 16;

    for (int k0 = 0; k0 < K; k0 += 32) {
        const int buf = (k0 >> 5) & 1;
        asm volatile("s_wait_asynccnt 0" ::: "memory");
        __syncthreads();                         // buffer `buf` ready for all waves

        if (k0 + 32 < K) stage(buf ^ 1, k0 + 32);  // overlap next DMA with compute

        // two A fragments (row tiles 0/1) from LDS
        const _Float16* sa0 = &sA[buf][lane & 15][0] + kg * 16;
        const _Float16* sa1 = &sA[buf][16 + (lane & 15)][0] + kg * 16;
        v8h a0lo = *(const v8h*)(sa0);
        v8h a0hi = *(const v8h*)(sa0 + 8);
        v8h a1lo = *(const v8h*)(sa1);
        v8h a1hi = *(const v8h*)(sa1 + 8);
        v16h a0, a1;
#pragma unroll
        for (int i = 0; i < 8; ++i) {
            a0[i] = a0lo[i]; a0[i + 8] = a0hi[i];
            a1[i] = a1lo[i]; a1[i + 8] = a1hi[i];
        }

        // batch all 4 B fragments (one clause), then 8 WMMAs
        v16h bf[4];
#pragma unroll
        for (int t = 0; t < 4; ++t) bf[t] = *(const v16h*)(pb + (size_t)t * 16 * K);
#pragma unroll
        for (int t = 0; t < 4; ++t) {
            acc[0][t] = __builtin_amdgcn_wmma_f32_16x16x32_f16(
                false, a0, false, bf[t], (short)0, acc[0][t], false, false);
            acc[1][t] = __builtin_amdgcn_wmma_f32_16x16x32_f16(
                false, a1, false, bf[t], (short)0, acc[1][t], false, false);
        }
        __syncthreads();                         // all waves consumed `buf`
        pb += 32;
        __builtin_prefetch(pb, 0, 1);            // global_prefetch_b8: next B slice
    }

#pragma unroll
    for (int rt = 0; rt < 2; ++rt) {
#pragma unroll
        for (int t = 0; t < 4; ++t) {
            int col = (tn0 + t) * 16 + (lane & 15);
            float bb = bias[col];
#pragma unroll
            for (int r = 0; r < 8; ++r) {
                int row = tm * 32 + rt * 16 + r + 8 * kg;
                float v = acc[rt][t][r] + bb;
                if (EPI == 1) v = gelu_tanh(v);
                if (EPI == 0) Yf[(size_t)row * Nc + col] = v;
                else          Yh[(size_t)row * Nc + col] = (_Float16)v;
            }
        }
    }
    (void)M;
}

// ---------------------------------------------------------------------------
// Kernel 3b: per-wave 16x16 WMMA GEMM for the two small GEMMs (Nc=64/32).
// ---------------------------------------------------------------------------
template <int EPI>
__global__ void __launch_bounds__(128)
wmma_gemm_small(const _Float16* __restrict__ X,
                const _Float16* __restrict__ W16,
                const float* __restrict__ bias,
                float* __restrict__ Yf,
                _Float16* __restrict__ Yh,
                int M, int Nc, int K) {
    int wave = blockIdx.x * (blockDim.x >> 5) + (threadIdx.x >> 5);
    int lane = threadIdx.x & 31;
    int tilesN = Nc >> 4;
    int tm = wave / tilesN;
    int tn = wave % tilesN;
    if (tm * 16 >= M) return;

    int kg = lane >> 4;
    const _Float16* pa = X + (size_t)(tm * 16 + (lane & 15)) * K + kg * 16;
    const _Float16* pb = W16 + (size_t)(tn * 16 + (lane & 15)) * K + kg * 16;

    v8f acc = {};
    for (int k0 = 0; k0 < K; k0 += 32) {
        v8h alo = *(const v8h*)(pa);
        v8h ahi = *(const v8h*)(pa + 8);
        v16h a;
#pragma unroll
        for (int i = 0; i < 8; ++i) { a[i] = alo[i]; a[i + 8] = ahi[i]; }
        v16h bf = *(const v16h*)(pb);
        acc = __builtin_amdgcn_wmma_f32_16x16x32_f16(
            false, a, false, bf, (short)0, acc, false, false);
        pa += 32;
        pb += 32;
    }

    int col = tn * 16 + (lane & 15);
    float bb = bias[col];
#pragma unroll
    for (int r = 0; r < 8; ++r) {
        int row = tm * 16 + r + 8 * kg;
        float v = acc[r] + bb;
        if (EPI == 1) v = gelu_tanh(v);
        if (EPI == 0) Yf[(size_t)row * Nc + col] = v;
        else          Yh[(size_t)row * Nc + col] = (_Float16)v;
    }
}

// ---------------------------------------------------------------------------
// Kernel 4: deformable bilinear sampling + attention-weighted accumulation.
// ---------------------------------------------------------------------------
__global__ void sample_kernel(const float* __restrict__ off,
                              const float* __restrict__ att,
                              const _Float16* __restrict__ vproj,
                              const float* __restrict__ ms,
                              const float* __restrict__ dtb,
                              _Float16* __restrict__ Xout) {
    int gid = blockIdx.x * blockDim.x + threadIdx.x;
    if (gid >= NPIX_Q * NH_) return;
    int h = gid & (NH_ - 1);
    int pos = gid >> 3;
    int bw = pos >> 6;
    int pix = pos & 63;
    int i = pix >> 3, j = pix & 7;
    int b = bw / NUMWIN_;
    int win = bw % NUMWIN_;

    float acc[HD_];
#pragma unroll
    for (int d = 0; d < HD_; ++d) acc[d] = 0.f;

    const float refx = (float)j * (1.0f / 7.0f);
    const float refy = (float)i * (1.0f / 7.0f);

    for (int n = 0; n < N_; ++n) {
        float msv = ms[(b * N_ + n) * NH_ + h];
        float db  = dtb[b * N_ + n];
        size_t vbase = (size_t)((b * N_ + n) * NUMWIN_ + win) << 6;
#pragma unroll
        for (int p = 0; p < NP_; ++p) {
            float ox = off[(size_t)pos * 64 + h * 8 + p * 2 + 0] * msv;
            float oy = off[(size_t)pos * 64 + h * 8 + p * 2 + 1] * msv;
            float tx = fminf(fmaxf(refx + ox, 0.f), 1.f);
            float ty = fminf(fmaxf(refy + oy, 0.f), 1.f);
            float x = fminf(fmaxf(8.0f * tx - 0.5f, 0.f), 7.0f);
            float y = fminf(fmaxf(8.0f * ty - 0.5f, 0.f), 7.0f);
            float x0f = floorf(x), y0f = floorf(y);
            float wx = x - x0f, wy = y - y0f;
            int x0 = (int)x0f, y0 = (int)y0f;
            int x1 = min(x0 + 1, 7), y1 = min(y0 + 1, 7);
            float aw = att[(size_t)pos * 32 + h * 4 + p] + db;
            float w00 = (1.f - wx) * (1.f - wy) * aw;
            float w01 = wx * (1.f - wy) * aw;
            float w10 = (1.f - wx) * wy * aw;
            float w11 = wx * wy * aw;
            const v8h* c00 = (const v8h*)(vproj + (vbase + (size_t)(y0 * 8 + x0)) * C_ + h * HD_);
            const v8h* c01 = (const v8h*)(vproj + (vbase + (size_t)(y0 * 8 + x1)) * C_ + h * HD_);
            const v8h* c10 = (const v8h*)(vproj + (vbase + (size_t)(y1 * 8 + x0)) * C_ + h * HD_);
            const v8h* c11 = (const v8h*)(vproj + (vbase + (size_t)(y1 * 8 + x1)) * C_ + h * HD_);
#pragma unroll
            for (int q = 0; q < HD_ / 8; ++q) {
                v8h t00 = c00[q], t01 = c01[q], t10 = c10[q], t11 = c11[q];
#pragma unroll
                for (int e = 0; e < 8; ++e) {
                    acc[q * 8 + e] += w00 * (float)t00[e] + w01 * (float)t01[e]
                                    + w10 * (float)t10[e] + w11 * (float)t11[e];
                }
            }
        }
    }
    _Float16* o = Xout + (size_t)pos * C_ + h * HD_;
#pragma unroll
    for (int d = 0; d < HD_; ++d) o[d] = (_Float16)acc[d];
}

// ---------------------------------------------------------------------------
// Kernel 5: softmax over the 128 logits -> entropy + confidence (d_out tail).
// ---------------------------------------------------------------------------
__global__ void softent_kernel(const float* __restrict__ att,
                               const float* __restrict__ dtb,
                               float* __restrict__ outp) {
    int pos = blockIdx.x * blockDim.x + threadIdx.x;
    if (pos >= NPIX_Q) return;
    int bw = pos >> 6, pix = pos & 63;
    int i = pix >> 3, j = pix & 7;
    int b = bw / NUMWIN_, win = bw % NUMWIN_;

    float av[32];
    float ma = -1e30f;
#pragma unroll
    for (int c = 0; c < 32; ++c) {
        av[c] = att[(size_t)pos * 32 + c];
        ma = fmaxf(ma, av[c]);
    }
    float dv[N_];
    float md = -1e30f;
#pragma unroll
    for (int n = 0; n < N_; ++n) { dv[n] = dtb[b * N_ + n]; md = fmaxf(md, dv[n]); }
    float Mx = ma + md;
    float S = 0.f;
#pragma unroll
    for (int n = 0; n < N_; ++n) {
        float en = __expf(dv[n] - md);
        float sc = 0.f;
#pragma unroll
        for (int c = 0; c < 32; ++c) sc += __expf(av[c] - ma);
        S += en * sc;
    }
    float inv = 1.0f / S;
    float ent = 0.f;
#pragma unroll
    for (int n = 0; n < N_; ++n) {
#pragma unroll
        for (int c = 0; c < 32; ++c) {
            float aa = __expf(av[c] + dv[n] - Mx) * inv;
            ent -= aa * __logf(aa + 1e-8f);
        }
    }
    float conf = inv;
    int hh = (win / HWIN_) * 8 + i;
    int ww = (win % HWIN_) * 8 + j;
    int idx = b * H_ * W_DIM + hh * W_DIM + ww;
    const size_t FUSED_N = (size_t)B_ * C_ * H_ * W_DIM;
    outp[FUSED_N + idx] = conf;
    outp[FUSED_N + NPIX_Q + idx] = ent;
}

// ---------------------------------------------------------------------------
// Kernel 6: gated residual: fused = query + a_q[b,c] * wrev(Yfused)
// ---------------------------------------------------------------------------
__global__ void fuse_kernel(const float* __restrict__ query,
                            const float* __restrict__ p_q,
                            const float* __restrict__ Yf,
                            float* __restrict__ outp) {
    size_t gid = (size_t)blockIdx.x * blockDim.x + threadIdx.x;
    const size_t TOT = (size_t)B_ * C_ * H_ * W_DIM;
    if (gid >= TOT) return;
    int b = (int)(gid / ((size_t)C_ * H_ * W_DIM));
    int rem = (int)(gid % ((size_t)C_ * H_ * W_DIM));
    int c = rem / (H_ * W_DIM);
    int pix = rem % (H_ * W_DIM);
    int h = pix / W_DIM, w = pix % W_DIM;
    int row = ((b * NUMWIN_ + (h >> 3) * HWIN_ + (w >> 3)) << 6) + ((h & 7) << 3) + (w & 7);
    float aq = p_q[b * 3 * C_ + 2 * C_ + c];
    outp[gid] = query[gid] + aq * Yf[(size_t)row * C_ + c];
}

// ---------------------------------------------------------------------------
extern "C" void kernel_launch(void* const* d_in, const int* in_sizes, int n_in,
                              void* d_out, int out_size, void* d_ws, size_t ws_size,
                              hipStream_t stream) {
    const float* query  = (const float*)d_in[0];
    const float* values = (const float*)d_in[1];
    const float* rel_dt = (const float*)d_in[2];
    const float* tgt    = (const float*)d_in[3];
    const float* W_q    = (const float*)d_in[4];
    const float* b_q    = (const float*)d_in[5];
    const float* W_kv   = (const float*)d_in[6];
    const float* b_kv   = (const float*)d_in[7];
    const float* W_o1   = (const float*)d_in[8];
    const float* b_o1   = (const float*)d_in[9];
    const float* W_o2   = (const float*)d_in[10];
    const float* b_o2   = (const float*)d_in[11];
    const float* W_a1   = (const float*)d_in[12];
    const float* b_a1   = (const float*)d_in[13];
    const float* W_a2   = (const float*)d_in[14];
    const float* b_a2   = (const float*)d_in[15];
    const float* W_v    = (const float*)d_in[16];
    const float* b_v    = (const float*)d_in[17];
    const float* W_out  = (const float*)d_in[18];
    const float* b_out  = (const float*)d_in[19];
    const float* W_ms   = (const float*)d_in[20];
    const float* b_ms   = (const float*)d_in[21];
    float* outp = (float*)d_out;

    // ---- workspace carving (aligned to 256B) ----
    char* wsp = (char*)d_ws;
    size_t off_b = 0;
    auto carve = [&](size_t bytes) -> char* {
        char* p = wsp + off_b;
        off_b += (bytes + 255) & ~(size_t)255;
        return p;
    };
    float*    p_q    = (float*)carve(B_ * 3 * C_ * 4);
    float*    p_kv   = (float*)carve(B_ * 3 * C_ * 4);
    float*    ms     = (float*)carve(B_ * N_ * NH_ * 4);
    float*    dtb    = (float*)carve(B_ * N_ * 4);
    _Float16* Wo1h   = (_Float16*)carve(128 * 256 * 2);
    _Float16* Wa1h   = (_Float16*)carve(128 * 256 * 2);
    _Float16* Wo2h   = (_Float16*)carve(64 * 128 * 2);
    _Float16* Wa2h   = (_Float16*)carve(32 * 128 * 2);
    _Float16* Wvh    = (_Float16*)carve(256 * 256 * 2);
    _Float16* Wouth  = (_Float16*)carve(256 * 256 * 2);
    _Float16* Xq     = (_Float16*)carve((size_t)NPIX_Q * C_ * 2);       // 9.4 MB
    _Float16* Xv     = (_Float16*)carve((size_t)NPIX_V * C_ * 2);       // 37.7 MB
    _Float16* vproj  = (_Float16*)carve((size_t)NPIX_V * C_ * 2);       // 37.7 MB
    _Float16* h1_off = (_Float16*)carve((size_t)NPIX_Q * 128 * 2);      // 4.7 MB
    _Float16* h1_att = (_Float16*)carve((size_t)NPIX_Q * 128 * 2);      // 4.7 MB
    float*    offb   = (float*)carve((size_t)NPIX_Q * 64 * 4);          // 4.7 MB
    float*    attb   = (float*)carve((size_t)NPIX_Q * 32 * 4);          // 2.4 MB
    _Float16* Xout   = (_Float16*)carve((size_t)NPIX_Q * C_ * 2);       // 9.4 MB
    float*    Yfused = (float*)carve((size_t)NPIX_Q * C_ * 4);          // 18.9 MB
    (void)ws_size; (void)in_sizes; (void)n_in; (void)out_size;

    // ---- 0: weight conversion to f16 ----
    cvt_w_kernel<<<(128 * 256 + 255) / 256, 256, 0, stream>>>(W_o1, Wo1h, 128 * 256);
    cvt_w_kernel<<<(128 * 256 + 255) / 256, 256, 0, stream>>>(W_a1, Wa1h, 128 * 256);
    cvt_w_kernel<<<(64 * 128 + 255) / 256, 256, 0, stream>>>(W_o2, Wo2h, 64 * 128);
    cvt_w_kernel<<<(32 * 128 + 255) / 256, 256, 0, stream>>>(W_a2, Wa2h, 32 * 128);
    cvt_w_kernel<<<(256 * 256 + 255) / 256, 256, 0, stream>>>(W_v, Wvh, 256 * 256);
    cvt_w_kernel<<<(256 * 256 + 255) / 256, 256, 0, stream>>>(W_out, Wouth, 256 * 256);

    // ---- 1: AdaLN params / motion scale ----
    params_kernel<<<(2 * B_ * 3 * C_ + 255) / 256, 256, 0, stream>>>(
        tgt, W_q, b_q, W_kv, b_kv, rel_dt, W_ms, b_ms, p_q, p_kv, ms, dtb);

    // ---- 2: AdaLN + window-partition to f16 activations ----
    adaln_q_kernel<<<(NPIX_Q + 255) / 256, 256, 0, stream>>>(query, p_q, Xq);
    adaln_v_kernel<<<(NPIX_V + 255) / 256, 256, 0, stream>>>(values, p_kv, Xv);

    // ---- 3: WMMA GEMMs ----
    // offset/attn net stage 1: [18432,256] x [128,256]^T, GELU -> f16 (NW=2)
    wmma_gemm_lds<2, 1><<<NPIX_Q / 32, 64, 0, stream>>>(
        Xq, Wo1h, b_o1, nullptr, h1_off, NPIX_Q, 128, C_);
    wmma_gemm_lds<2, 1><<<NPIX_Q / 32, 64, 0, stream>>>(
        Xq, Wa1h, b_a1, nullptr, h1_att, NPIX_Q, 128, C_);
    // offset net stage 2: [18432,128] x [64,128]^T -> f32
    wmma_gemm_small<0><<<((NPIX_Q / 16) * 4 + 3) / 4, 128, 0, stream>>>(
        h1_off, Wo2h, b_o2, offb, nullptr, NPIX_Q, 64, 128);
    // attn net stage 2: [18432,128] x [32,128]^T -> f32
    wmma_gemm_small<0><<<((NPIX_Q / 16) * 2 + 3) / 4, 128, 0, stream>>>(
        h1_att, Wa2h, b_a2, attb, nullptr, NPIX_Q, 32, 128);
    // value projection: [73728,256] x [256,256]^T -> f16 (dominant GEMM, NW=4)
    wmma_gemm_lds<4, 2><<<NPIX_V / 32, 128, 0, stream>>>(
        Xv, Wvh, b_v, nullptr, vproj, NPIX_V, C_, C_);

    // ---- 4: deformable sampling + weighted accumulation -> f16 Xout ----
    sample_kernel<<<(NPIX_Q * NH_ + 255) / 256, 256, 0, stream>>>(
        offb, attb, vproj, ms, dtb, Xout);

    // ---- 5: softmax entropy / confidence -> d_out tail ----
    softent_kernel<<<(NPIX_Q + 255) / 256, 256, 0, stream>>>(attb, dtb, outp);

    // ---- 6: W_out GEMM: [18432,256] x [256,256]^T -> f32 (NW=4) ----
    wmma_gemm_lds<4, 0><<<NPIX_Q / 32, 128, 0, stream>>>(
        Xout, Wouth, b_out, Yfused, nullptr, NPIX_Q, C_, C_);

    // ---- 7: gated residual -> d_out head ----
    fuse_kernel<<<(int)(((size_t)B_ * C_ * H_ * W_DIM + 255) / 256), 256, 0, stream>>>(
        query, p_q, Yfused, outp);
}
